// CrossAttention_4518305595787
// MI455X (gfx1250) — compile-verified, matrix operand order checked by reading, not run on hardware
//
#include <hip/hip_runtime.h>

typedef __attribute__((ext_vector_type(16))) _Float16 v16h;
typedef __attribute__((ext_vector_type(8)))  _Float16 v8h;
typedef __attribute__((ext_vector_type(8)))  float    v8f;
typedef __attribute__((ext_vector_type(4)))  float    v4f;
typedef __attribute__((ext_vector_type(4)))  unsigned int u32x4;
typedef __attribute__((ext_vector_type(8)))  int      i32x8;
typedef __attribute__((ext_vector_type(4)))  int      i32x4;

#define DIMN   1024
#define NHEADS 16
#define HD     64
#define SEQ    2048
#define BATCH  4
#define TOKENS (BATCH * SEQ)   // 8192
#define ATT_SCALE 0.125f       // HEAD_DIM^-0.5 = 1/8

// ---------------------------------------------------------------------------
// TDM: DMA a 2D f16 tile (tile_h rows x tile_w elements, row stride `stride`
// elements) from global memory into LDS at byte offset lds_off, laid out
// densely row-major [tile_h][tile_w]. Descriptor per CDNA5 ISA ch.8:
//   group0: count=1 | lds_addr | global_addr(57b) | type=2
//   group1: data_size=2B, tensor_dim0/1, tile_dim0/1/2, dim strides
//   remaining groups: zero (2D tensor, iterate/gather off)
// Issue from ONE wave (EXEC ignored by TDM); completion via TENSORcnt.
// ---------------------------------------------------------------------------
__device__ __forceinline__ void tdm_load_2d_f16(const void* gsrc,
                                                unsigned int lds_off,
                                                unsigned int tensor_w,
                                                unsigned int tensor_h,
                                                unsigned int stride,
                                                unsigned int tile_w,
                                                unsigned int tile_h) {
  const unsigned long long ga = (unsigned long long)(uintptr_t)gsrc;
  u32x4 g0;
  g0[0] = 1u;                                             // count=1, user mode
  g0[1] = lds_off;                                        // LDS byte address
  g0[2] = (unsigned int)(ga & 0xffffffffu);               // global_addr[31:0]
  g0[3] = (unsigned int)((ga >> 32) & 0x01ffffffu)        // global_addr[56:32]
          | (2u << 30);                                   // type=2 ("image")
  i32x8 g1;
  g1[0] = (int)(1u << 16);                                // data_size=1 (2B)
  g1[1] = (int)((tensor_w & 0xffffu) << 16);              // tensor_dim0 lo
  g1[2] = (int)((tensor_w >> 16) | ((tensor_h & 0xffffu) << 16));
  g1[3] = (int)((tensor_h >> 16) | (tile_w << 16));       // tile_dim0
  g1[4] = (int)(tile_h & 0xffffu);                        // tile_dim1 (dim2=0)
  g1[5] = (int)stride;                                    // dim0_stride[31:0]
  g1[6] = 0;                                              // dim0_stride hi, dim1_stride lo
  g1[7] = 0;                                              // dim1_stride hi
  const i32x4 gz4 = {0, 0, 0, 0};
  const i32x8 gz8 = {0, 0, 0, 0, 0, 0, 0, 0};
  __builtin_amdgcn_tensor_load_to_lds(g0, g1, gz4, gz4, gz8, 0);
}

__device__ __forceinline__ unsigned int lds_offset_of(const void* p) {
  // Generic pointer to LDS = {SHARED_BASE_hi32, lds_byte_offset} (aperture
  // mapping, ISA 10.2): low 32 bits are the wave-relative LDS address.
  return (unsigned int)(uintptr_t)p;
}

// ---------------------------------------------------------------------------
// Load a 16x32 f16 tile (row-major in LDS, leading dim `ld` halfs) into the
// CDNA5 WMMA A/B fragment layout (ISA 7.12.2, 16-bit A 16x32):
//   lane L: row = L&15, K-chunks kb..kb+7 and kb+16..kb+23, kb=(L>>4)*8.
// ---------------------------------------------------------------------------
__device__ __forceinline__ v16h ldsFrag(const _Float16* base, int ld) {
  const int lane = threadIdx.x & 31;
  const int r    = lane & 15;
  const int kb   = (lane >> 4) << 3;
  const _Float16* p = base + r * ld + kb;
  const v8h lo = *(const v8h*)(p);
  const v8h hi = *(const v8h*)(p + 16);
  v16h f;
#pragma unroll
  for (int i = 0; i < 8; ++i) { f[i] = lo[i]; f[8 + i] = hi[i]; }
  return f;
}

// Row reductions across the 16 columns of a C-tile row (one 16-lane half).
__device__ __forceinline__ float rowMax(float v) {
#pragma unroll
  for (int m = 8; m >= 1; m >>= 1) v = fmaxf(v, __shfl_xor(v, m, 32));
  return v;
}
__device__ __forceinline__ float rowSum(float v) {
#pragma unroll
  for (int m = 8; m >= 1; m >>= 1) v += __shfl_xor(v, m, 32);
  return v;
}

// ---------------------------------------------------------------------------
// Kernel 1: q/k projection.  C[8192, 2048] = X @ Wqkv[0:2048,:]^T + bqkv.
// f32 sources -> VALU convert staging (no TDM possible), WMMA f16/f32-acc.
// ---------------------------------------------------------------------------
__global__ __launch_bounds__(128)
void qk_proj_kernel(const float* __restrict__ x,
                    const float* __restrict__ Wqkv,
                    const float* __restrict__ bqkv,
                    _Float16* __restrict__ q_ws,
                    _Float16* __restrict__ k_ws) {
  __shared__ __align__(16) _Float16 Xs[64][32];
  __shared__ __align__(16) _Float16 Ws[64][32];

  const int m0   = blockIdx.y * 64;
  const int n0   = blockIdx.x * 64;
  const int t    = threadIdx.x;
  const int wv   = t >> 5;
  const int lane = t & 31;

  v8f acc[4] = {};

  for (int k0 = 0; k0 < DIMN; k0 += 32) {
    __syncthreads();
    {   // stage X tile (f32 -> f16)
      const int row = t >> 1, half = (t & 1) * 16;
      const float* src = x + (size_t)(m0 + row) * DIMN + k0 + half;
      _Float16* dst = &Xs[row][half];
#pragma unroll
      for (int i = 0; i < 16; i += 4) {
        v4f f = *(const v4f*)(src + i);
        dst[i]   = (_Float16)f[0]; dst[i+1] = (_Float16)f[1];
        dst[i+2] = (_Float16)f[2]; dst[i+3] = (_Float16)f[3];
      }
    }
    {   // stage W tile
      const int row = t >> 1, half = (t & 1) * 16;
      const float* src = Wqkv + (size_t)(n0 + row) * DIMN + k0 + half;
      _Float16* dst = &Ws[row][half];
#pragma unroll
      for (int i = 0; i < 16; i += 4) {
        v4f f = *(const v4f*)(src + i);
        dst[i]   = (_Float16)f[0]; dst[i+1] = (_Float16)f[1];
        dst[i+2] = (_Float16)f[2]; dst[i+3] = (_Float16)f[3];
      }
    }
    __syncthreads();

    const v16h a = ldsFrag(&Xs[wv * 16][0], 32);
#pragma unroll
    for (int j = 0; j < 4; ++j) {
      const v16h b = ldsFrag(&Ws[j * 16][0], 32);
      acc[j] = __builtin_amdgcn_wmma_f32_16x16x32_f16(
          false, a, false, b, (short)0, acc[j], false, false);
    }
  }

  const int rowOff = (lane >> 4) * 8;
  const int col    = lane & 15;
#pragma unroll
  for (int j = 0; j < 4; ++j) {
    const int gcol = n0 + j * 16 + col;
    const float bias = bqkv[gcol];
#pragma unroll
    for (int r = 0; r < 8; ++r) {
      const int tok = m0 + wv * 16 + rowOff + r;
      const int b   = tok >> 11, n = tok & (SEQ - 1);
      const _Float16 hv = (_Float16)(acc[j][r] + bias);
      if (gcol < DIMN) {
        const int h = gcol >> 6, d = gcol & 63;
        q_ws[(((size_t)(b * NHEADS + h)) * SEQ + n) * HD + d] = hv;
      } else {
        const int c = gcol - DIMN;
        const int h = c >> 6, d = c & 63;
        k_ws[(((size_t)(b * NHEADS + h)) * SEQ + n) * HD + d] = hv;
      }
    }
  }
}

// ---------------------------------------------------------------------------
// Kernel 2: flash-style attention per (b,h), 64 queries/WG, 64-key tiles.
// Q and K tiles arrive via TDM (tensor_load_to_lds, TENSORcnt); V^T staged
// with VALU f32->f16 transpose overlapping the K-tile DMA.
// ---------------------------------------------------------------------------
__global__ __launch_bounds__(128)
void attn_kernel(const _Float16* __restrict__ q_ws,
                 const _Float16* __restrict__ k_ws,
                 const float* __restrict__ context,
                 _Float16* __restrict__ o_ws) {
  __shared__ __align__(16) _Float16 Qs[64][64];
  __shared__ __align__(16) _Float16 Ks[64][64];
  __shared__ __align__(16) _Float16 VTs[64][64];
  __shared__ __align__(16) _Float16 Ps[4][16 * 64];

  const int q0   = blockIdx.x * 64;
  const int bh   = blockIdx.y;
  const int b    = bh >> 4, h = bh & 15;
  const int t    = threadIdx.x;
  const int wv   = t >> 5;
  const int lane = t & 31;

  const _Float16* qbase = q_ws + (size_t)bh * SEQ * HD;
  const _Float16* kbase = k_ws + (size_t)bh * SEQ * HD;

  // Q tile via TDM (one wave issues; EXEC ignored by TDM)
  if (wv == 0) {
    tdm_load_2d_f16(qbase + (size_t)q0 * HD, lds_offset_of(&Qs[0][0]),
                    HD, SEQ, HD, 64, 64);
    __builtin_amdgcn_s_wait_tensorcnt(0);
  }
  __syncthreads();

  v16h aq[2];
  aq[0] = ldsFrag(&Qs[wv * 16][0],  64);
  aq[1] = ldsFrag(&Qs[wv * 16][32], 64);

  float mrun[8], lrun[8];
  v8f o[4] = {};
#pragma unroll
  for (int r = 0; r < 8; ++r) { mrun[r] = -1e30f; lrun[r] = 0.f; }

  for (int kt = 0; kt < SEQ / 64; ++kt) {
    __syncthreads();
    // K tile DMA: issue first so it overlaps the V^T convert staging below.
    if (wv == 0)
      tdm_load_2d_f16(kbase + (size_t)kt * 64 * HD, lds_offset_of(&Ks[0][0]),
                      HD, SEQ, HD, 64, 64);
    {   // stage V^T from context (f32 -> f16, transposed)
      const int row = t >> 1;          // key index within tile
      const int dh  = (t & 1) * 32;    // d offset
      const float* src = context + ((size_t)b * SEQ + kt * 64 + row) * DIMN + h * HD + dh;
#pragma unroll
      for (int d = 0; d < 32; ++d) VTs[dh + d][row] = (_Float16)src[d];
    }
    if (wv == 0) __builtin_amdgcn_s_wait_tensorcnt(0);
    __syncthreads();

    // S = Q K^T (16 queries x 64 keys per wave)
    v8f s4[4];
#pragma unroll
    for (int j = 0; j < 4; ++j) {
      v8f sc = {};
#pragma unroll
      for (int ks = 0; ks < 2; ++ks) {
        const v16h bk = ldsFrag(&Ks[j * 16][ks * 32], 64);
        sc = __builtin_amdgcn_wmma_f32_16x16x32_f16(
            false, aq[ks], false, bk, (short)0, sc, false, false);
      }
      s4[j] = sc;
    }

    // Online softmax
#pragma unroll
    for (int r = 0; r < 8; ++r) {
      float tm = -1e30f;
#pragma unroll
      for (int j = 0; j < 4; ++j) { s4[j][r] *= ATT_SCALE; tm = fmaxf(tm, s4[j][r]); }
      tm = rowMax(tm);
      const float mnew  = fmaxf(mrun[r], tm);
      const float alpha = __expf(mrun[r] - mnew);
      float rs = 0.f;
#pragma unroll
      for (int j = 0; j < 4; ++j) {
        const float e = __expf(s4[j][r] - mnew);
        s4[j][r] = e; rs += e;
      }
      rs = rowSum(rs);
      lrun[r] = lrun[r] * alpha + rs;
      mrun[r] = mnew;
#pragma unroll
      for (int j = 0; j < 4; ++j) o[j][r] *= alpha;
    }

    // P: accumulator layout -> per-wave LDS region -> A fragment
    _Float16* pw = &Ps[wv][0];
    {
      const int rowOff = (lane >> 4) * 8, col = lane & 15;
#pragma unroll
      for (int j = 0; j < 4; ++j)
#pragma unroll
        for (int r = 0; r < 8; ++r)
          pw[(rowOff + r) * 64 + j * 16 + col] = (_Float16)s4[j][r];
    }
    // Wave-private region: wave's own DS stores complete is sufficient.
    asm volatile("s_wait_dscnt 0x0" ::: "memory");

    // O += P V
#pragma unroll
    for (int ks = 0; ks < 2; ++ks) {
      const v16h pa = ldsFrag(pw + ks * 32, 64);
#pragma unroll
      for (int j = 0; j < 4; ++j) {
        const v16h bv = ldsFrag(&VTs[j * 16][ks * 32], 64);
        o[j] = __builtin_amdgcn_wmma_f32_16x16x32_f16(
            false, pa, false, bv, (short)0, o[j], false, false);
      }
    }
  }

  // Epilogue: normalize, store f16 [token, dim] for the out-proj GEMM
  const int rowOff = (lane >> 4) * 8, col = lane & 15;
#pragma unroll
  for (int r = 0; r < 8; ++r) {
    const float inv = 1.f / lrun[r];
    const int tok = b * SEQ + q0 + wv * 16 + rowOff + r;
#pragma unroll
    for (int j = 0; j < 4; ++j)
      o_ws[(size_t)tok * DIMN + h * HD + j * 16 + col] = (_Float16)(o[j][r] * inv);
  }
}

// ---------------------------------------------------------------------------
// Kernel 3: output projection. out[8192,1024] = A(f16) @ Wout^T + bout (f32).
// A tile arrives via TDM overlapping the Wout f32->f16 convert staging.
// ---------------------------------------------------------------------------
__global__ __launch_bounds__(128)
void out_proj_kernel(const _Float16* __restrict__ a_ws,
                     const float* __restrict__ Wout,
                     const float* __restrict__ bout,
                     float* __restrict__ out) {
  __shared__ __align__(16) _Float16 As[64][32];
  __shared__ __align__(16) _Float16 Ws[64][32];

  const int m0   = blockIdx.y * 64;
  const int n0   = blockIdx.x * 64;
  const int t    = threadIdx.x;
  const int wv   = t >> 5;
  const int lane = t & 31;

  v8f acc[4] = {};

  for (int k0 = 0; k0 < DIMN; k0 += 32) {
    __syncthreads();
    // A tile via TDM: 64 rows x 32 halfs, row stride DIMN
    if (wv == 0)
      tdm_load_2d_f16(a_ws + (size_t)m0 * DIMN + k0, lds_offset_of(&As[0][0]),
                      DIMN, TOKENS, DIMN, 32, 64);
    {   // stage Wout tile (f32 -> f16) — overlaps the DMA
      const int row = t >> 1, half = (t & 1) * 16;
      const float* src = Wout + (size_t)(n0 + row) * DIMN + k0 + half;
      _Float16* dst = &Ws[row][half];
#pragma unroll
      for (int i = 0; i < 16; i += 4) {
        v4f f = *(const v4f*)(src + i);
        dst[i]   = (_Float16)f[0]; dst[i+1] = (_Float16)f[1];
        dst[i+2] = (_Float16)f[2]; dst[i+3] = (_Float16)f[3];
      }
    }
    if (wv == 0) __builtin_amdgcn_s_wait_tensorcnt(0);
    __syncthreads();

    const v16h a = ldsFrag(&As[wv * 16][0], 32);
#pragma unroll
    for (int j = 0; j < 4; ++j) {
      const v16h bw = ldsFrag(&Ws[j * 16][0], 32);
      acc[j] = __builtin_amdgcn_wmma_f32_16x16x32_f16(
          false, a, false, bw, (short)0, acc[j], false, false);
    }
  }

  const int rowOff = (lane >> 4) * 8, col = lane & 15;
#pragma unroll
  for (int j = 0; j < 4; ++j) {
    const float bias = bout[n0 + j * 16 + col];
#pragma unroll
    for (int r = 0; r < 8; ++r) {
      const int tok = m0 + wv * 16 + rowOff + r;
      out[(size_t)tok * DIMN + n0 + j * 16 + col] = acc[j][r] + bias;
    }
  }
}

// ---------------------------------------------------------------------------
extern "C" void kernel_launch(void* const* d_in, const int* in_sizes, int n_in,
                              void* d_out, int out_size, void* d_ws, size_t ws_size,
                              hipStream_t stream) {
  (void)in_sizes; (void)n_in; (void)out_size; (void)ws_size;

  const float* x       = (const float*)d_in[0];
  const float* context = (const float*)d_in[1];
  const float* Wqkv    = (const float*)d_in[2];
  const float* bqkv    = (const float*)d_in[3];
  const float* Wout    = (const float*)d_in[4];
  const float* bout    = (const float*)d_in[5];
  float* out = (float*)d_out;

  _Float16* q_ws = (_Float16*)d_ws;                    // 16 MB
  _Float16* k_ws = q_ws + (size_t)TOKENS * DIMN;       // 16 MB
  _Float16* o_ws = k_ws + (size_t)TOKENS * DIMN;       // 16 MB

  const dim3 blk(128, 1, 1);
  qk_proj_kernel <<<dim3(2048 / 64, TOKENS / 64, 1), blk, 0, stream>>>(
      x, Wqkv, bqkv, q_ws, k_ws);
  attn_kernel    <<<dim3(SEQ / 64, BATCH * NHEADS, 1), blk, 0, stream>>>(
      q_ws, k_ws, context, o_ws);
  out_proj_kernel<<<dim3(DIMN / 64, TOKENS / 64, 1), blk, 0, stream>>>(
      o_ws, Wout, bout, out);
}
